// TransformerBlock_33990371181214
// MI455X (gfx1250) — matrix operand order repaired and checked by reading, not compiled, not run on hardware
//
#include <hip/hip_runtime.h>
#include <stdint.h>

// ---------------------------------------------------------------------------
// Types & helpers
// ---------------------------------------------------------------------------
typedef __bf16 bf16_t;
typedef __attribute__((ext_vector_type(16))) __bf16 v16bf;
typedef __attribute__((ext_vector_type(8))) float v8f;
typedef __attribute__((ext_vector_type(4))) int v4i_async;

union Frag16 { uint4 q[2]; v16bf v; };
union FragAcc { v8f v; float f[8]; };
union FU { float f; uint32_t u; };

// CDNA5 async global->LDS copy (ASYNCcnt-tracked), with fallback.
#if defined(__gfx1250__) && __has_builtin(__builtin_amdgcn_global_load_async_to_lds_b128)
#define USE_ASYNC_LDS 1
#else
#define USE_ASYNC_LDS 0
#endif

__device__ __forceinline__ void async_cp16(const void* gsrc, void* ldst) {
#if USE_ASYNC_LDS
  __builtin_amdgcn_global_load_async_to_lds_b128(
      (__attribute__((address_space(1))) v4i_async*)(uintptr_t)gsrc,
      (__attribute__((address_space(3))) v4i_async*)(uint32_t)(uintptr_t)ldst,
      0, 0);
#else
  *(uint4*)ldst = *(const uint4*)gsrc;
#endif
}

__device__ __forceinline__ void async_wait0() {
#if USE_ASYNC_LDS
#if __has_builtin(__builtin_amdgcn_s_wait_asynccnt)
  __builtin_amdgcn_s_wait_asynccnt(0);
#else
  asm volatile("s_wait_asynccnt 0" ::: "memory");
#endif
#endif
}

__device__ __forceinline__ uint16_t f2bf(float x) {
  FU t; t.f = x;
  uint32_t r = t.u + 0x7FFFu + ((t.u >> 16) & 1u);   // round-nearest-even
  return (uint16_t)(r >> 16);
}

__device__ __forceinline__ v8f wmma_bf16(v16bf a, v16bf b, v8f c) {
  // D(f32 16x16) = A(16x32 bf16) * B(32x16 bf16) + C
  return __builtin_amdgcn_wmma_f32_16x16x32_bf16(false, a, false, b, (short)0, c,
                                                 false, false);
}

__device__ __forceinline__ float gelu_tanh(float x) {
  float c = 0.7978845608028654f * (x + 0.044715f * x * x * x);
  return 0.5f * x * (1.0f + tanhf(c));
}

// ---------------------------------------------------------------------------
// fp32 -> bf16 transpose:  out[c][r] = bf16(in[r][c]),  in is R x C
// ---------------------------------------------------------------------------
__global__ __launch_bounds__(256) void transpose_bf16_kernel(
    const float* __restrict__ in, uint16_t* __restrict__ out, int R, int C) {
  __shared__ uint16_t t[32][33];
  const int r0 = blockIdx.y * 32, c0 = blockIdx.x * 32;
  const int tr = threadIdx.x >> 3;          // 0..31
  const int tc = (threadIdx.x & 7) << 2;    // 0,4,..,28
  float4 v = *(const float4*)&in[(size_t)(r0 + tr) * C + c0 + tc];
  t[tc + 0][tr] = f2bf(v.x);
  t[tc + 1][tr] = f2bf(v.y);
  t[tc + 2][tr] = f2bf(v.z);
  t[tc + 3][tr] = f2bf(v.w);
  __syncthreads();
  union { uint16_t s[4]; uint2 u; } pk;
  pk.s[0] = t[tr][tc + 0];
  pk.s[1] = t[tr][tc + 1];
  pk.s[2] = t[tr][tc + 2];
  pk.s[3] = t[tr][tc + 3];
  *(uint2*)&out[(size_t)(c0 + tr) * R + r0 + tc] = pk.u;
}

// ---------------------------------------------------------------------------
// LayerNorm over D=2048, one block (256 thr) per row, bf16 output
// ---------------------------------------------------------------------------
__global__ __launch_bounds__(256) void layernorm_bf16_kernel(
    const float* __restrict__ x, const float* __restrict__ g,
    const float* __restrict__ sft, uint16_t* __restrict__ out, int D) {
  __shared__ float red[256];
  const int row = blockIdx.x, tid = threadIdx.x;
  const float* xr = x + (size_t)row * D;
  const int c0 = tid * 8;
  float v[8];
  float4 a = *(const float4*)&xr[c0];
  float4 b = *(const float4*)&xr[c0 + 4];
  v[0]=a.x; v[1]=a.y; v[2]=a.z; v[3]=a.w; v[4]=b.x; v[5]=b.y; v[6]=b.z; v[7]=b.w;
  float s = 0.f, sq = 0.f;
  #pragma unroll
  for (int i = 0; i < 8; ++i) { s += v[i]; sq += v[i] * v[i]; }
  red[tid] = s; __syncthreads();
  for (int o = 128; o > 0; o >>= 1) { if (tid < o) red[tid] += red[tid + o]; __syncthreads(); }
  const float mean = red[0] / (float)D;
  __syncthreads();
  red[tid] = sq; __syncthreads();
  for (int o = 128; o > 0; o >>= 1) { if (tid < o) red[tid] += red[tid + o]; __syncthreads(); }
  const float var = red[0] / (float)D - mean * mean;
  const float rstd = rsqrtf(var + 1e-5f);
  uint16_t* orow = out + (size_t)row * D;
  #pragma unroll
  for (int i = 0; i < 8; ++i) {
    int c = c0 + i;
    orow[c] = f2bf((v[i] - mean) * rstd * g[c] + sft[c]);
  }
}

// ---------------------------------------------------------------------------
// WMMA bf16 GEMM: C[M,N] = A[M,K] * Bt[N,K]^T  (+epilogue)
//   EPI 1: bf16 out, scattered to [b,h,s,hd] head-major layout
//   EPI 2: f32 out = acc + bias[col] + resid[row*N+col]
//   EPI 3: bf16 out = gelu(acc + bias[col])
// Block tile 128x128, K-step 32, 8 waves (2x4 frags each),
// double-buffered LDS with async global->LDS staging.
// ---------------------------------------------------------------------------
template <int EPI>
__global__ __launch_bounds__(256) void gemm_bf16_kernel(
    const uint16_t* __restrict__ A, const uint16_t* __restrict__ Bt,
    void* __restrict__ Out, const float* __restrict__ bias,
    const float* __restrict__ resid, int M, int N, int K, int S, int HD) {
  __shared__ uint16_t la[2][128][40];   // 32 K + 8 pad (80B rows, 16B aligned)
  __shared__ uint16_t lb[2][128][40];
  const int tid = threadIdx.x;
  const int wave = tid >> 5, lane = tid & 31;
  const int wm = wave >> 1, wn = wave & 1;
  const int half = lane >> 4, l16 = lane & 15;
  const int m0 = blockIdx.y * 128, n0 = blockIdx.x * 128;

  FragAcc acc[2][4];
  #pragma unroll
  for (int i = 0; i < 2; ++i)
    #pragma unroll
    for (int j = 0; j < 4; ++j)
      #pragma unroll
      for (int r = 0; r < 8; ++r) acc[i][j].f[r] = 0.0f;

  auto stage = [&](int buf, int k0) {
    for (int c = tid; c < 512; c += 256) {
      const int row = c >> 2, kc = (c & 3) << 3;
      async_cp16(&A[(size_t)(m0 + row) * K + k0 + kc], &la[buf][row][kc]);
      async_cp16(&Bt[(size_t)(n0 + row) * K + k0 + kc], &lb[buf][row][kc]);
    }
  };

  stage(0, 0);
  async_wait0();
  __syncthreads();

  int buf = 0;
  for (int k0 = 0; k0 < K; k0 += 32) {
    const int k1 = k0 + 32;
    if (k1 < K) stage(buf ^ 1, k1);   // overlap next slab with compute

    Frag16 af[2], bfr[4];
    #pragma unroll
    for (int mt = 0; mt < 2; ++mt) {
      const uint16_t* p = &la[buf][wm * 32 + mt * 16 + l16][0];
      af[mt].q[0] = *(const uint4*)(p + half * 8);        // K: h*8..h*8+7
      af[mt].q[1] = *(const uint4*)(p + 16 + half * 8);   // K: 16+h*8..+7
    }
    #pragma unroll
    for (int nt = 0; nt < 4; ++nt) {
      const uint16_t* p = &lb[buf][wn * 64 + nt * 16 + l16][0];
      bfr[nt].q[0] = *(const uint4*)(p + half * 8);       // K: h*8..h*8+7
      bfr[nt].q[1] = *(const uint4*)(p + 16 + half * 8);  // K: 16+h*8..+7
    }
    #pragma unroll
    for (int mt = 0; mt < 2; ++mt)
      #pragma unroll
      for (int nt = 0; nt < 4; ++nt)
        acc[mt][nt].v = wmma_bf16(af[mt].v, bfr[nt].v, acc[mt][nt].v);

    async_wait0();       // this wave's staged copies landed in LDS
    __syncthreads();     // all waves done reading buf & staging buf^1
    buf ^= 1;
  }

  // epilogue: D-frag element (r): row = r + 8*half, col = l16
  #pragma unroll
  for (int mt = 0; mt < 2; ++mt)
    #pragma unroll
    for (int nt = 0; nt < 4; ++nt)
      #pragma unroll
      for (int r = 0; r < 8; ++r) {
        const int row = m0 + wm * 32 + mt * 16 + r + 8 * half;
        const int col = n0 + wn * 64 + nt * 16 + l16;
        const float va = acc[mt][nt].f[r];
        if (EPI == 1) {
          const int Hn = N / HD;
          const int b = row / S, sr = row % S, hh = col / HD, hd = col % HD;
          ((uint16_t*)Out)[(((size_t)b * Hn + hh) * S + sr) * HD + hd] = f2bf(va);
        } else if (EPI == 2) {
          const size_t idx = (size_t)row * N + col;
          ((float*)Out)[idx] = va + bias[col] + resid[idx];
        } else {  // EPI == 3
          ((uint16_t*)Out)[(size_t)row * N + col] = f2bf(gelu_tanh(va + bias[col]));
        }
      }
}

// ---------------------------------------------------------------------------
// Flash attention (causal): Q,K,V bf16 in [b*h][S][128]; ctx bf16 [b*s][D]
// Block: 128 q-rows (8 waves x 16 rows), streams 64-key tiles via LDS.
// K-tile staged with async copies; V transposed through VGPRs.
// ---------------------------------------------------------------------------
__global__ __launch_bounds__(256) void attn_kernel(
    const uint16_t* __restrict__ Q, const uint16_t* __restrict__ Kb,
    const uint16_t* __restrict__ Vb, uint16_t* __restrict__ ctx,
    int S, int Hn, float scale, int Dfull) {
  const int HD = 128;
  __shared__ uint16_t lk[64][136];     // [key][hd], padded
  __shared__ uint16_t lvt[128][72];    // [hd][key], padded (V transposed)
  __shared__ uint16_t lp[8][16][72];   // per-wave P strips [row][key]

  const int tid = threadIdx.x, wave = tid >> 5, lane = tid & 31;
  const int half = lane >> 4, l16 = lane & 15;
  const int qt = blockIdx.x, bh = blockIdx.y;
  const int b = bh / Hn, h = bh % Hn;
  const size_t base = (size_t)bh * S * HD;

  // Q fragments held in registers for all 4 K-dim steps (HD=128)
  const int qrow = qt * 128 + wave * 16 + l16;
  Frag16 qf[4];
  #pragma unroll
  for (int kk = 0; kk < 4; ++kk) {
    const uint16_t* p = &Q[base + (size_t)qrow * HD + kk * 32];
    qf[kk].q[0] = *(const uint4*)(p + half * 8);
    qf[kk].q[1] = *(const uint4*)(p + 16 + half * 8);
  }

  float m_row[8], l_row[8];
  FragAcc o[8];
  #pragma unroll
  for (int r = 0; r < 8; ++r) { m_row[r] = -1e30f; l_row[r] = 0.0f; }
  #pragma unroll
  for (int ot = 0; ot < 8; ++ot)
    #pragma unroll
    for (int r = 0; r < 8; ++r) o[ot].f[r] = 0.0f;

  const int wave_min_row = qt * 128 + wave * 16;
  const int wave_max_row = wave_min_row + 15;
  const int jmax = 2 * qt + 1;

  for (int j = 0; j <= jmax; ++j) {
    // stage K tile (async, row-major) and V tile (transposed) into LDS
    for (int c = tid; c < 1024; c += 256) {
      const int kr = c >> 4, kc = (c & 15) << 3;
      async_cp16(&Kb[base + (size_t)(j * 64 + kr) * HD + kc], &lk[kr][kc]);
    }
    for (int c = tid; c < 1024; c += 256) {
      const int vr = c >> 4, vc = (c & 15) << 3;
      union { uint4 u; uint16_t s[8]; } tmp;
      tmp.u = *(const uint4*)&Vb[base + (size_t)(j * 64 + vr) * HD + vc];
      #pragma unroll
      for (int i = 0; i < 8; ++i) lvt[vc + i][vr] = tmp.s[i];
    }
    async_wait0();
    __syncthreads();

    const bool active = (j * 64) <= wave_max_row;       // wave-uniform
    if (active) {
      const bool diag = (j * 64 + 63) > wave_min_row;
      // scores S = Q * K^T  (4 n-tiles of 16 keys)
      FragAcc sc[4];
      #pragma unroll
      for (int nt = 0; nt < 4; ++nt) {
        #pragma unroll
        for (int r = 0; r < 8; ++r) sc[nt].f[r] = 0.0f;
        #pragma unroll
        for (int kk = 0; kk < 4; ++kk) {
          Frag16 bfrag;
          const uint16_t* p = &lk[nt * 16 + l16][kk * 32];
          bfrag.q[0] = *(const uint4*)(p + half * 8);
          bfrag.q[1] = *(const uint4*)(p + 16 + half * 8);
          sc[nt].v = wmma_bf16(qf[kk].v, bfrag.v, sc[nt].v);
        }
      }
      // scale + causal mask + row max
      float tmax[8];
      #pragma unroll
      for (int r = 0; r < 8; ++r) tmax[r] = -1e30f;
      #pragma unroll
      for (int nt = 0; nt < 4; ++nt)
        #pragma unroll
        for (int r = 0; r < 8; ++r) {
          float sv = sc[nt].f[r] * scale;
          if (diag) {
            const int grow = wave_min_row + r + 8 * half;
            const int gcol = j * 64 + nt * 16 + l16;
            if (gcol > grow) sv = -1e30f;
          }
          sc[nt].f[r] = sv;
          tmax[r] = fmaxf(tmax[r], sv);
        }
      #pragma unroll
      for (int off = 1; off < 16; off <<= 1)
        #pragma unroll
        for (int r = 0; r < 8; ++r)
          tmax[r] = fmaxf(tmax[r], __shfl_xor(tmax[r], off, 32));
      // online softmax update
      float corr[8], m_new[8], lsum[8];
      #pragma unroll
      for (int r = 0; r < 8; ++r) {
        m_new[r] = fmaxf(m_row[r], tmax[r]);
        corr[r] = __expf(m_row[r] - m_new[r]);
        m_row[r] = m_new[r];
        lsum[r] = 0.0f;
      }
      #pragma unroll
      for (int nt = 0; nt < 4; ++nt)
        #pragma unroll
        for (int r = 0; r < 8; ++r) {
          const float p = __expf(sc[nt].f[r] - m_new[r]);
          sc[nt].f[r] = p;
          lsum[r] += p;
        }
      #pragma unroll
      for (int off = 1; off < 16; off <<= 1)
        #pragma unroll
        for (int r = 0; r < 8; ++r) lsum[r] += __shfl_xor(lsum[r], off, 32);
      #pragma unroll
      for (int r = 0; r < 8; ++r) l_row[r] = l_row[r] * corr[r] + lsum[r];
      #pragma unroll
      for (int ot = 0; ot < 8; ++ot)
        #pragma unroll
        for (int r = 0; r < 8; ++r) o[ot].f[r] *= corr[r];
      // stash P (bf16) to this wave's private LDS strip
      #pragma unroll
      for (int nt = 0; nt < 4; ++nt)
        #pragma unroll
        for (int r = 0; r < 8; ++r)
          lp[wave][r + 8 * half][nt * 16 + l16] = f2bf(sc[nt].f[r]);
      // O += P * V   (A-frags from P strip; B-frags from transposed V)
      #pragma unroll
      for (int kk = 0; kk < 2; ++kk) {
        Frag16 pf;
        const uint16_t* pp = &lp[wave][l16][kk * 32];
        pf.q[0] = *(const uint4*)(pp + half * 8);
        pf.q[1] = *(const uint4*)(pp + 16 + half * 8);
        #pragma unroll
        for (int ot = 0; ot < 8; ++ot) {
          Frag16 vf;
          const uint16_t* vp = &lvt[ot * 16 + l16][kk * 32];
          vf.q[0] = *(const uint4*)(vp + half * 8);
          vf.q[1] = *(const uint4*)(vp + 16 + half * 8);
          o[ot].v = wmma_bf16(pf.v, vf.v, o[ot].v);
        }
      }
    }
    __syncthreads();
  }

  // finalize: divide by row sums, scatter ctx to [b*s][D] layout
  #pragma unroll
  for (int ot = 0; ot < 8; ++ot)
    #pragma unroll
    for (int r = 0; r < 8; ++r) {
      const float inv = 1.0f / l_row[r];
      const int row = wave_min_row + r + 8 * half;
      const int col = h * HD + ot * 16 + l16;
      ctx[(size_t)(b * S + row) * Dfull + col] = f2bf(o[ot].f[r] * inv);
    }
}

// ---------------------------------------------------------------------------
// Launcher
// ---------------------------------------------------------------------------
extern "C" void kernel_launch(void* const* d_in, const int* in_sizes, int n_in,
                              void* d_out, int out_size, void* d_ws, size_t ws_size,
                              hipStream_t stream) {
  (void)in_sizes; (void)n_in; (void)out_size; (void)ws_size;
  constexpr int D = 2048, Hn = 16, HD = 128, B = 4, S = 2048;
  constexpr int M = B * S;              // 8192
  constexpr int DF = 4 * D;             // 8192

  const float* x  = (const float*)d_in[0];
  const float* Wq = (const float*)d_in[1];
  const float* Wk = (const float*)d_in[2];
  const float* Wv = (const float*)d_in[3];
  const float* Wo = (const float*)d_in[4];
  const float* bo = (const float*)d_in[5];
  const float* W1 = (const float*)d_in[6];
  const float* b1 = (const float*)d_in[7];
  const float* W2 = (const float*)d_in[8];
  const float* b2 = (const float*)d_in[9];
  const float* g1 = (const float*)d_in[10];
  const float* s1 = (const float*)d_in[11];
  const float* g2 = (const float*)d_in[12];
  const float* s2 = (const float*)d_in[13];

  // workspace carving (bf16 = uint16_t elements)
  uint16_t* Wtq = (uint16_t*)d_ws;                       // [D][D]  (W^T)
  uint16_t* Wtk = Wtq + (size_t)D * D;
  uint16_t* Wtv = Wtk + (size_t)D * D;
  uint16_t* Wto = Wtv + (size_t)D * D;
  uint16_t* Wt1 = Wto + (size_t)D * D;                   // [4D][D]
  uint16_t* Wt2 = Wt1 + (size_t)DF * D;                  // [D][4D]
  uint16_t* hb  = Wt2 + (size_t)DF * D;                  // [M][D] (LN out, reused)
  uint16_t* qb  = hb  + (size_t)M * D;                   // [b*h][S][HD]
  uint16_t* kb  = qb  + (size_t)M * D;
  uint16_t* vb  = kb  + (size_t)M * D;
  uint16_t* cb  = vb  + (size_t)M * D;                   // ctx [M][D]
  uint16_t* gb  = qb;                                    // FFN hidden [M][4D] aliases q/k/v/ctx

  // 1) weight convert+transpose to bf16
  transpose_bf16_kernel<<<dim3(D / 32, D / 32), 256, 0, stream>>>(Wq, Wtq, D, D);
  transpose_bf16_kernel<<<dim3(D / 32, D / 32), 256, 0, stream>>>(Wk, Wtk, D, D);
  transpose_bf16_kernel<<<dim3(D / 32, D / 32), 256, 0, stream>>>(Wv, Wtv, D, D);
  transpose_bf16_kernel<<<dim3(D / 32, D / 32), 256, 0, stream>>>(Wo, Wto, D, D);
  transpose_bf16_kernel<<<dim3(DF / 32, D / 32), 256, 0, stream>>>(W1, Wt1, D, DF);
  transpose_bf16_kernel<<<dim3(D / 32, DF / 32), 256, 0, stream>>>(W2, Wt2, DF, D);

  // 2) LN1
  layernorm_bf16_kernel<<<M, 256, 0, stream>>>(x, g1, s1, hb, D);

  // 3) QKV projections (head-major outputs)
  gemm_bf16_kernel<1><<<dim3(D / 128, M / 128), 256, 0, stream>>>(hb, Wtq, qb, nullptr, nullptr, M, D, D, S, HD);
  gemm_bf16_kernel<1><<<dim3(D / 128, M / 128), 256, 0, stream>>>(hb, Wtk, kb, nullptr, nullptr, M, D, D, S, HD);
  gemm_bf16_kernel<1><<<dim3(D / 128, M / 128), 256, 0, stream>>>(hb, Wtv, vb, nullptr, nullptr, M, D, D, S, HD);

  // 4) causal flash attention
  attn_kernel<<<dim3(S / 128, B * Hn), 256, 0, stream>>>(qb, kb, vb, cb, S, Hn,
                                                         0.08838834764831845f, D);

  // 5) output projection + residual (fp32 into d_out)
  gemm_bf16_kernel<2><<<dim3(D / 128, M / 128), 256, 0, stream>>>(cb, Wto, d_out, bo, x, M, D, D, S, HD);

  // 6) LN2 + FFN
  layernorm_bf16_kernel<<<M, 256, 0, stream>>>((const float*)d_out, g2, s2, hb, D);
  gemm_bf16_kernel<3><<<dim3(DF / 128, M / 128), 256, 0, stream>>>(hb, Wt1, gb, b1, nullptr, M, DF, D, S, HD);
  gemm_bf16_kernel<2><<<dim3(D / 128, M / 128), 256, 0, stream>>>(gb, Wt2, d_out, b2, (const float*)d_out, M, D, DF, S, HD);
}
